// Tokenizer_9053791060293
// MI455X (gfx1250) — matrix operand order repaired and checked by reading, not compiled, not run on hardware
//
#include <hip/hip_runtime.h>

typedef __attribute__((ext_vector_type(16))) _Float16 v16h;
typedef __attribute__((ext_vector_type(8)))  float    v8f;

#define B_ 16
#define T_ 4096
#define C_ 64
#define KC 1024          // padded code count (codes 1..1023 of codebook + 1 pad)
#define NCODES 1023

// d_out layout (floats), tuple return order
#define O_ZQ     2
#define O_NZQ    (2 + B_*T_*C_)
#define O_NMASK  (2 + 2*B_*T_*C_)
#define O_SEL    (2 + 2*B_*T_*C_ + B_*T_)

// ---------------- K1: row L2-normalize (one wave per row) ----------------
__global__ void k_normalize(const float* __restrict__ z, float* __restrict__ zn) {
    int w = (blockIdx.x * blockDim.x + threadIdx.x) >> 5;   // row id
    int lane = threadIdx.x & 31;
    if (w >= B_ * T_) return;
    const float* zr = z + (size_t)w * C_;
    float a = zr[lane], b = zr[lane + 32];
    float ss = a * a + b * b;
    #pragma unroll
    for (int m = 16; m; m >>= 1) ss += __shfl_xor(ss, m, 32);
    float sc = 1.0f / fmaxf(sqrtf(ss), 1e-12f);
    float* o = zn + (size_t)w * C_;
    o[lane] = a * sc;
    o[lane + 32] = b * sc;
}

// ------------- K2: codebook -> f16 (rows 1..1023) + fp32 |e|^2 -------------
__global__ void k_prep_codebook(const float* __restrict__ cw,
                                _Float16* __restrict__ eh,
                                float* __restrict__ enorm) {
    int j = blockIdx.x;            // 0..1023 (padded index into e)
    int lane = threadIdx.x;        // 0..31
    if (j < NCODES) {
        const float* er = cw + (size_t)(j + 1) * C_;
        float a = er[lane], b = er[lane + 32];
        eh[(size_t)j * C_ + lane]      = (_Float16)a;
        eh[(size_t)j * C_ + lane + 32] = (_Float16)b;
        float ss = a * a + b * b;
        #pragma unroll
        for (int m = 16; m; m >>= 1) ss += __shfl_xor(ss, m, 32);
        if (lane == 0) enorm[j] = ss;
    } else {
        eh[(size_t)j * C_ + lane]      = (_Float16)0.0f;
        eh[(size_t)j * C_ + lane + 32] = (_Float16)0.0f;
        if (lane == 0) enorm[j] = 3.0e38f;   // pad code never wins
    }
}

// ---------- K3: WMMA distance GEMM + fused argmin (1 wave / 16 rows) ----------
__global__ void __launch_bounds__(32)
k_argmin_wmma(const float* __restrict__ zn, const _Float16* __restrict__ eh,
              const float* __restrict__ enorm, int* __restrict__ idxArr) {
    const int rowBase = blockIdx.x * 16;
    const int lane = threadIdx.x;
    const int half = lane >> 4;
    const int lm   = lane & 15;

    // A operand: row (rowBase+lm), pre-scaled by -2 so D = -2*z.e
    // A k order chosen here: lanes<16 hold K {0..15}+kb, lanes>=16 hold {16..31}+kb (contiguous, matches B)
    const float* zr = zn + (size_t)(rowBase + lm) * C_;
    v16h a0, a1;
    {
        const int ra = half * 16;
        const int rb = half * 16 + 8;
        #pragma unroll
        for (int i = 0; i < 8; ++i) {
            a0[i]     = (_Float16)(-2.0f * zr[ra + i]);
            a0[8 + i] = (_Float16)(-2.0f * zr[rb + i]);
            a1[i]     = (_Float16)(-2.0f * zr[32 + ra + i]);
            a1[8 + i] = (_Float16)(-2.0f * zr[32 + rb + i]);
        }
    }

    float bs[8];
    int   bj[8];
    #pragma unroll
    for (int i = 0; i < 8; ++i) { bs[i] = 3.4e38f; bj[i] = 0; }

    for (int t = 0; t < KC / 16; ++t) {
        const int j = t * 16 + lm;                 // this lane's code column
        // B 32x16 f16, k order chosen here: lane holds column n=lm, K = kb + half*16 .. +15 (contiguous)
        const _Float16* ebase = eh + (size_t)j * C_ + half * 16;
        v16h b0 = *(const v16h*)(ebase);
        v16h b1 = *(const v16h*)(ebase + 32);
        float en = enorm[j];
        if (t + 1 < KC / 16)
            __builtin_prefetch((const void*)(ebase + 16 * C_), 0, 0);

        v8f c = {};
        c = __builtin_amdgcn_wmma_f32_16x16x32_f16(false, a0, false, b0,
                                                   (short)0, c, false, false);
        c = __builtin_amdgcn_wmma_f32_16x16x32_f16(false, a1, false, b1,
                                                   (short)0, c, false, false);
        #pragma unroll
        for (int i = 0; i < 8; ++i) {
            float s = en + c[i];                   // |e|^2 - 2 z.e
            if (s < bs[i]) { bs[i] = s; bj[i] = j; }   // j increases with t -> first-min kept
        }
    }

    // cross-lane argmin within each 16-lane half; ties -> lowest code index
    #pragma unroll
    for (int m = 8; m; m >>= 1) {
        #pragma unroll
        for (int i = 0; i < 8; ++i) {
            float s2 = __shfl_xor(bs[i], m, 16);
            int   j2 = __shfl_xor(bj[i], m, 16);
            if (s2 < bs[i] || (s2 == bs[i] && j2 < bj[i])) { bs[i] = s2; bj[i] = j2; }
        }
    }
    // D layout: lanes 0-15 slot i = row i; lanes 16-31 slot i = row 8+i
    if (lane == 0) {
        #pragma unroll
        for (int i = 0; i < 8; ++i) idxArr[rowBase + i] = bj[i];
    } else if (lane == 16) {
        #pragma unroll
        for (int i = 0; i < 8; ++i) idxArr[rowBase + 8 + i] = bj[i];
    }
}

// ------- K4: z_q = e[idx]*mask (overwrites zn in-place) + commit partials -------
__global__ void k_zq_commit(const float* __restrict__ cw, const int* __restrict__ idxArr,
                            const float* __restrict__ mask, float* __restrict__ zq,
                            float* __restrict__ commitPart) {
    int w = (blockIdx.x * blockDim.x + threadIdx.x) >> 5;
    int lane = threadIdx.x & 31;
    if (w >= B_ * T_) return;
    int j = idxArr[w];
    float m = mask[w];
    const float* er = cw + (size_t)(j + 1) * C_;
    float* zr = zq + (size_t)w * C_;
    float z0 = zr[lane], z1 = zr[lane + 32];       // currently holds normalized z
    float q0 = er[lane] * m, q1 = er[lane + 32] * m;
    float d0 = z0 - q0, d1 = z1 - q1;
    float ss = (d0 * d0 + d1 * d1) * m;
    #pragma unroll
    for (int mm = 16; mm; mm >>= 1) ss += __shfl_xor(ss, mm, 32);
    zr[lane] = q0;
    zr[lane + 32] = q1;
    if (lane == 0) commitPart[w] = ss;
}

// ---------------- K5: smoothness partials per (b, t) pair ----------------
__global__ void k_smooth(const float* __restrict__ zq, const float* __restrict__ mask,
                         float* __restrict__ smoothPart, float* __restrict__ smoothMaskPart) {
    int w = (blockIdx.x * blockDim.x + threadIdx.x) >> 5;
    int lane = threadIdx.x & 31;
    const int per = T_ - 1;
    if (w >= B_ * per) return;
    int b = w / per, t = w % per;
    size_t r0 = ((size_t)b * T_ + t) * C_;
    float m = mask[(size_t)b * T_ + t + 1];
    float d0 = zq[r0 + lane]      - zq[r0 + C_ + lane];
    float d1 = zq[r0 + lane + 32] - zq[r0 + C_ + lane + 32];
    float ss = (d0 * d0 + d1 * d1) * m;
    #pragma unroll
    for (int mm = 16; mm; mm >>= 1) ss += __shfl_xor(ss, mm, 32);
    if (lane == 0) { smoothPart[w] = ss; smoothMaskPart[w] = m; }
}

// ------------- deterministic single-block sum reduction -------------
__global__ void k_reduce(const float* __restrict__ in, int count, float* __restrict__ out) {
    __shared__ float sh[256];
    float acc = 0.0f;
    for (int i = threadIdx.x; i < count; i += 256) acc += in[i];
    sh[threadIdx.x] = acc;
    __syncthreads();
    for (int s = 128; s; s >>= 1) {
        if ((int)threadIdx.x < s) sh[threadIdx.x] += sh[threadIdx.x + s];
        __syncthreads();
    }
    if (threadIdx.x == 0) out[0] = sh[0];
}

// ------------- K6: per-batch stream-compaction of change points -------------
__global__ void k_compact(const int* __restrict__ idxArr, const float* __restrict__ mask,
                          int* __restrict__ posArr, int* __restrict__ nArr) {
    int b = blockIdx.x;
    int td = threadIdx.x;                 // 0..255
    const int per = T_ / 256;             // 16
    __shared__ int offs[257];
    const int base = td * per;
    const int* ib = idxArr + (size_t)b * T_;
    const float* mb = mask + (size_t)b * T_;
    int cnt = 0;
    for (int i = 0; i < per; ++i) {
        int t = base + i;
        bool st = (t == 0) || ((ib[t] != ib[t - 1]) && (mb[t] > 0.0f));
        cnt += st ? 1 : 0;
    }
    offs[td + 1] = cnt;
    __syncthreads();
    if (td == 0) {
        offs[0] = 0;
        for (int k = 1; k <= 256; ++k) offs[k] += offs[k - 1];
        nArr[b] = offs[256];
    }
    __syncthreads();
    int o = offs[td];
    int* pb = posArr + (size_t)b * T_;
    for (int i = 0; i < per; ++i) {
        int t = base + i;
        bool st = (t == 0) || ((ib[t] != ib[t - 1]) && (mb[t] > 0.0f));
        if (st) pb[o++] = t;
    }
}

// ------------- K7: gather n_z_q / n_mask / selected_encodings -------------
__global__ void k_gather(const float* __restrict__ zq, const int* __restrict__ idxArr,
                         const int* __restrict__ posArr, const int* __restrict__ nArr,
                         float* __restrict__ nzq, float* __restrict__ nmask,
                         float* __restrict__ sel) {
    int blk = blockIdx.x;                 // b*T + s
    int b = blk >> 12;                    // T_ = 4096
    int s = blk & (T_ - 1);
    int tid = threadIdx.x;                // 0..63
    float* orow = nzq + (size_t)blk * C_;
    if (s < nArr[b]) {
        int pos = posArr[(size_t)b * T_ + s];
        size_t src = ((size_t)b * T_ + pos) * C_;
        orow[tid] = zq[src + tid];
        if (tid == 0) {
            nmask[blk] = 1.0f;
            sel[blk] = (float)(idxArr[(size_t)b * T_ + pos] + 1);
        }
    } else {
        orow[tid] = 0.0f;
        if (tid == 0) { nmask[blk] = 0.0f; sel[blk] = 0.0f; }
    }
}

// ---------------- K8: final loss scalars ----------------
__global__ void k_finalize(const float* __restrict__ sums, float* __restrict__ out) {
    if (threadIdx.x == 0) {
        out[0] = sums[2] / sums[3];   // smoothness_loss
        out[1] = sums[0] / sums[1];   // commitment_loss
    }
}

extern "C" void kernel_launch(void* const* d_in, const int* in_sizes, int n_in,
                              void* d_out, int out_size, void* d_ws, size_t ws_size,
                              hipStream_t stream) {
    (void)in_sizes; (void)n_in; (void)out_size; (void)ws_size;
    const float* z    = (const float*)d_in[0];
    const float* cw   = (const float*)d_in[1];
    const float* mask = (const float*)d_in[2];
    float* out = (float*)d_out;

    char* ws = (char*)d_ws;
    _Float16* eh        = (_Float16*)(ws);              // 131072 B
    float*    enorm     = (float*)(ws + 131072);        //   4096 B
    int*      idxArr    = (int*)  (ws + 135168);        // 262144 B
    int*      posArr    = (int*)  (ws + 397312);        // 262144 B
    int*      nArr      = (int*)  (ws + 659456);        //    256 B
    float*    sums      = (float*)(ws + 659712);        //    256 B
    float*    commitP   = (float*)(ws + 659968);        // 262144 B
    float*    smoothP   = (float*)(ws + 922112);        // 262144 B
    float*    smoothMP  = (float*)(ws + 1184256);       // 262144 B (~1.45 MB total)

    float* zq = out + O_ZQ;   // staged: first holds normalized z, then z_q

    k_normalize    <<<(B_ * T_) / 8, 256, 0, stream>>>(z, zq);
    k_prep_codebook<<<KC, 32, 0, stream>>>(cw, eh, enorm);
    k_argmin_wmma  <<<(B_ * T_) / 16, 32, 0, stream>>>(zq, eh, enorm, idxArr);
    k_zq_commit    <<<(B_ * T_) / 8, 256, 0, stream>>>(cw, idxArr, mask, zq, commitP);
    k_smooth       <<<(B_ * (T_ - 1) + 7) / 8, 256, 0, stream>>>(zq, mask, smoothP, smoothMP);

    k_reduce<<<1, 256, 0, stream>>>(commitP,  B_ * T_,       &sums[0]);
    k_reduce<<<1, 256, 0, stream>>>(mask,     B_ * T_,       &sums[1]);
    k_reduce<<<1, 256, 0, stream>>>(smoothP,  B_ * (T_ - 1), &sums[2]);
    k_reduce<<<1, 256, 0, stream>>>(smoothMP, B_ * (T_ - 1), &sums[3]);

    k_compact <<<B_, 256, 0, stream>>>(idxArr, mask, posArr, nArr);
    k_gather  <<<B_ * T_, 64, 0, stream>>>(zq, idxArr, posArr, nArr,
                                           out + O_NZQ, out + O_NMASK, out + O_SEL);
    k_finalize<<<1, 32, 0, stream>>>(sums, out);
}
